// WindowAttention_local_57423712747939
// MI455X (gfx1250) — matrix operand order repaired and checked by reading, not compiled, not run on hardware
//
#include <hip/hip_runtime.h>
#include <hip/hip_bf16.h>

// ---------------------------------------------------------------------------
// Swin window attention, single fused WMMA kernel per window (gfx1250 wave32).
//   B=8, H=W=128, C=256, window 8x8 (64 tokens), 8 heads, head-dim 32.
//
// All WMMA operands (weights, activations, Q/K/V, P, O, bias) are kept in the
// exact per-lane fragment layout so fragment traffic is pure b128 vectors:
//   prologue: swizzle_w_kernel    fp32 weights -> bf16 B-fragment stream
//             swizzle_bias_kernel rpb[rel_idx] -> f32 C-tile fragment stream
//   main    : fused_win_kernel    QKV GEMM -> attention -> proj, all
//             intermediates resident in 160 KB of LDS.
//
// Fragment maps (CDNA5 ISA 7.12.2, bf16 16x16x32):
//   A (16Mx32K): lane = (m&15) + 16*((k>>3)&1) ; elem j = ((k>>4)<<3)|(k&7)
//   B (32Kx16N): lane = (n&15) + 16*((k>>3)&1) ; elem j = ((k>>4)<<3)|(k&7)
//   C/D       : lane = (n&15) + 16*(m>=8)      ; vgpr r = m&7  (f32 x8)
// ---------------------------------------------------------------------------

typedef __attribute__((ext_vector_type(16))) __bf16 v16bf;
typedef __attribute__((ext_vector_type(8)))  float  v8f;

namespace {
constexpr int kWS    = 8;
constexpr int kHeads = 8;
constexpr int kDim   = 256;
constexpr int kNTok  = 64;
constexpr int kNWin  = 2048;            // 8 * 16 * 16 windows
constexpr int kFrag  = 512;             // elements per fragment (32 lanes x 16)
}

__device__ __forceinline__ v8f wmma_bf16(v16bf a, v16bf b, v8f c) {
  return __builtin_amdgcn_wmma_f32_16x16x32_bf16(
      false, a, false, b, (short)0, c, false, false);
}

// Contiguous per-lane fragment load (32 bytes/lane -> 2 x b128).
__device__ __forceinline__ v16bf ld_frag(const __bf16* base) {
  return *(const v16bf*)(base + (threadIdx.x & 31) * 16);
}

// ---------------------------------------------------------------------------
// Weight pre-swizzle: W fp32 [N][K] row-major -> bf16 B-fragment stream.
// ---------------------------------------------------------------------------
__global__ __launch_bounds__(256) void swizzle_w_kernel(
    const float* __restrict__ W, __bf16* __restrict__ Wsw, int N, int K) {
  const int e = blockIdx.x * 256 + threadIdx.x;
  if (e >= N * K) return;
  const int j    = e & 15;
  const int lane = (e >> 4) & 31;
  const int kt   = K >> 5;
  const int ks   = (e >> 9) % kt;
  const int nt   = (e >> 9) / kt;
  const int k    = (ks << 5) + ((j >> 3) << 4) + ((lane >> 4) << 3) + (j & 7);
  const int n    = (nt << 4) + (lane & 15);
  Wsw[e] = (__bf16)W[(size_t)n * K + k];
}

// ---------------------------------------------------------------------------
// Bias pre-swizzle: bias[h][m][n] = rpb[rel_idx[m,n]*8 + h] stored as C-tile
// fragments: bsw[(((h*4+mt)*4+nt)*32 + lane)*8 + r]  (8 f32 contiguous/lane).
// ---------------------------------------------------------------------------
__global__ __launch_bounds__(256) void swizzle_bias_kernel(
    const float* __restrict__ rpb, const int* __restrict__ rel_idx,
    float* __restrict__ bsw) {
  const int e = blockIdx.x * 256 + threadIdx.x;
  if (e >= kHeads * 4 * 4 * 32 * 8) return;
  const int r    = e & 7;
  const int lane = (e >> 3) & 31;
  const int nt   = (e >> 8) & 3;
  const int mt   = (e >> 10) & 3;
  const int h    = e >> 12;
  const int m = mt * 16 + r + ((lane >> 4) << 3);
  const int n = nt * 16 + (lane & 15);
  bsw[e] = rpb[rel_idx[m * kNTok + n] * kHeads + h];
}

// ---------------------------------------------------------------------------
// Fused per-window kernel. grid = 2048 windows, block = 256 threads (8 waves).
// Dynamic LDS (163840 B): xs | qs | ks | vs | os, 32 KB each.
// ---------------------------------------------------------------------------
__global__ __launch_bounds__(256) void fused_win_kernel(
    const float* __restrict__ x, const __bf16* __restrict__ wqkv_sw,
    const float* __restrict__ qkv_b, const __bf16* __restrict__ wproj_sw,
    const float* __restrict__ proj_b, const float* __restrict__ bias_sw,
    float* __restrict__ out) {
  extern __shared__ __bf16 smem[];
  __bf16* xs  = smem;                    // x (A-swizzled); reused for P strips
  __bf16* qs  = smem + 16384;            // Q, A-swizzled per head
  __bf16* ksm = smem + 32768;            // K, B-swizzled per head (B = K^T)
  __bf16* vsm = smem + 49152;            // V, B-swizzled per head
  __bf16* osm = smem + 65536;            // attn out, A-swizzled for proj

  const int w    = blockIdx.x;
  const int b    = w >> 8;
  const int wh   = (w >> 4) & 15;
  const int ww   = w & 15;
  const int tid  = threadIdx.x;
  const int wvu  = __builtin_amdgcn_readfirstlane(tid >> 5);  // scalar wave id
  const int lane = tid & 31;
  const int half = lane >> 4;
  const int lm   = lane & 15;

  // ---- Phase A: stage x window (64x256 fp32) into LDS, A-swizzled bf16 ----
  {
    const int c4  = tid & 63;            // float4 index within token (const)
    const int t0  = tid >> 6;            // 0..3
    const int kst = c4 >> 3;
    const int lnx = ((c4 >> 1) & 1) << 4;
    const int jb  = ((c4 & 4) << 1) + ((c4 & 1) << 2);
    const float* xb =
        x + ((size_t)(b * 128 + wh * kWS) * 128 + ww * kWS) * kDim + c4 * 4;
#pragma unroll
    for (int it = 0; it < 16; ++it) {
      const int t = t0 + it * 4;         // token 0..63
      const float4 f =
          *(const float4*)(xb + (size_t)((t >> 3) * 128 + (t & 7)) * kDim);
      union { __bf16 h[4]; uint2 u; } tmp;
      tmp.h[0] = (__bf16)f.x; tmp.h[1] = (__bf16)f.y;
      tmp.h[2] = (__bf16)f.z; tmp.h[3] = (__bf16)f.w;
      *(uint2*)(xs + ((t >> 4) * 8 + kst) * kFrag + ((t & 15) + lnx) * 16 +
                jb) = tmp.u;
    }
  }
  __syncthreads();

  // ---- Phase B: QKV GEMM (64x256 @ 256x768) -> Q/K/V swizzled in LDS ----
  const float scale = 0.17677669529663687f;  // 1/sqrt(32), folded into Q
  for (int nt = wvu; nt < 48; nt += 8) {     // scalar loop index
    const __bf16* wp = wqkv_sw + (size_t)(nt * 8) * kFrag;
    __builtin_prefetch(wp, 0, 1);
    v16bf bw[8];
#pragma unroll
    for (int ks = 0; ks < 8; ++ks) bw[ks] = ld_frag(wp + ks * kFrag);

    // sel/head/d_hi are scalar functions of nt; d-low is the lane column.
    const int sel  = nt >> 4;            // 0=q 1=k 2=v
    const int head = (nt & 15) >> 1;
    const int dhi  = nt & 1;             // d>>4
    const float bias = qkv_b[nt * 16 + lm];

    v8f acc[4] = {};                     // 4 independent accumulator chains
#pragma unroll
    for (int ks = 0; ks < 8; ++ks) {
#pragma unroll
      for (int mt = 0; mt < 4; ++mt)
        acc[mt] = wmma_bf16(ld_frag(xs + (mt * 8 + ks) * kFrag), bw[ks],
                            acc[mt]);
    }

    if (sel == 0) {                      // Q -> A-swizzled (scalar branch)
      __bf16* qd = qs + head * 2048 +
                   (((lm >> 3) & 1) << 4) * 16 + (half << 7) + dhi * 8 +
                   (lm & 7);
#pragma unroll
      for (int mt = 0; mt < 4; ++mt)
#pragma unroll
        for (int r = 0; r < 8; ++r)
          qd[mt * kFrag + r * 16] = (__bf16)((acc[mt][r] + bias) * scale);
    } else if (sel == 1) {               // K -> B-swizzled (lane = token n,
                                         // elem j = table-order d).
      __bf16* kd = ksm + head * 2048 +
                   (((lm >> 3) & 1) << 4) * 16 + (half << 7) + dhi * 8 +
                   (lm & 7);
#pragma unroll
      for (int mt = 0; mt < 4; ++mt)
#pragma unroll
        for (int r = 0; r < 8; ++r)
          kd[mt * kFrag + r * 16] = (__bf16)(acc[mt][r] + bias);
    } else {                             // V -> B-swizzled
      __bf16* vd = vsm + head * 2048 + dhi * kFrag + (half << 8) + lm * 16;
#pragma unroll
      for (int mt = 0; mt < 4; ++mt)
#pragma unroll
        for (int r = 0; r < 8; ++r)
          vd[(mt >> 1) * 1024 + (mt & 1) * 8 + r] =
              (__bf16)(acc[mt][r] + bias);
    }
  }
  __syncthreads();

  // ---- Phase C: attention. 32 tasks = (head, 16-row strip), 4 per wave ----
  __bf16* psw = xs + wvu * 1024;         // per-wave P strip (A-swizzled)
#pragma unroll 1
  for (int it = 0; it < 4; ++it) {
    const int task = wvu + it * 8;       // scalar
    const int h  = task >> 2;
    const int mt = task & 3;

    // S = (Q*scale) K^T : head-dim 32 = one WMMA k-step per tile.
    const v16bf aq = ld_frag(qs + (h * 4 + mt) * kFrag);
    v8f s[4];
#pragma unroll
    for (int nt = 0; nt < 4; ++nt) {
      v8f z = {};
      s[nt] = wmma_bf16(aq, ld_frag(ksm + (h * 4 + nt) * kFrag), z);
      // bias tile as one 32-byte per-lane fragment load
      const v8f bb = *(const v8f*)(
          bias_sw + ((((h * 4 + mt) * 4 + nt) * 32) + lane) * 8);
      s[nt] = s[nt] + bb;
    }

    // Row softmax: C-tile rows live across one 16-lane half (wave32).
#pragma unroll
    for (int r = 0; r < 8; ++r) {
      float mx = s[0][r];
#pragma unroll
      for (int nt = 1; nt < 4; ++nt) mx = fmaxf(mx, s[nt][r]);
#pragma unroll
      for (int off = 1; off < 16; off <<= 1)
        mx = fmaxf(mx, __shfl_xor(mx, off, 32));
      float sum = 0.f;
#pragma unroll
      for (int nt = 0; nt < 4; ++nt) {
        s[nt][r] = __expf(s[nt][r] - mx);
        sum += s[nt][r];
      }
#pragma unroll
      for (int off = 1; off < 16; off <<= 1) sum += __shfl_xor(sum, off, 32);
      const float inv = 1.0f / sum;
      const int prow = r + (half << 3);
#pragma unroll
      for (int nt = 0; nt < 4; ++nt)     // store P, A-swizzled
        psw[(nt >> 1) * kFrag +
            (prow + (((lm >> 3) & 1) << 4)) * 16 + ((nt & 1) << 3) +
            (lm & 7)] = (__bf16)(s[nt][r] * inv);
    }

    // O = P @ V : K = 64 tokens -> 2 k-steps; two 16-wide d tiles.
    v8f o0 = {}, o1 = {};
#pragma unroll
    for (int ks = 0; ks < 2; ++ks) {
      const v16bf ap = ld_frag(psw + ks * kFrag);
      o0 = wmma_bf16(ap, ld_frag(vsm + (h * 4 + ks * 2 + 0) * kFrag), o0);
      o1 = wmma_bf16(ap, ld_frag(vsm + (h * 4 + ks * 2 + 1) * kFrag), o1);
    }
    // Store O into osm, A-swizzled for the proj GEMM (k-step index == h).
#pragma unroll
    for (int r = 0; r < 8; ++r) {
      __bf16* dst = osm + (mt * 8 + h) * kFrag +
                    (r + (half << 3) + (((lm >> 3) & 1) << 4)) * 16;
      dst[(lm & 7)]     = (__bf16)o0[r];
      dst[8 + (lm & 7)] = (__bf16)o1[r];
    }
  }
  __syncthreads();

  // ---- Phase D: proj (64x256 @ 256x256) + bias + window reverse ----
  for (int nt = wvu; nt < 16; nt += 8) {
    const __bf16* wp = wproj_sw + (size_t)(nt * 8) * kFrag;
    __builtin_prefetch(wp, 0, 1);
    v16bf bw[8];
#pragma unroll
    for (int ks = 0; ks < 8; ++ks) bw[ks] = ld_frag(wp + ks * kFrag);
    const int n = nt * 16 + lm;
    const float bias = proj_b[n];

    v8f acc[4] = {};
#pragma unroll
    for (int ks = 0; ks < 8; ++ks) {
#pragma unroll
      for (int mt = 0; mt < 4; ++mt)
        acc[mt] = wmma_bf16(ld_frag(osm + (mt * 8 + ks) * kFrag), bw[ks],
                            acc[mt]);
    }
    float* ob = out + ((size_t)b * 16384 + (wh * kWS) * 128 + ww * kWS) * kDim + n;
#pragma unroll
    for (int mt = 0; mt < 4; ++mt) {
      const int gr0 = (mt * 2 + half) * 128;     // (m>>3)*128, m=mt*16+8*half+r
#pragma unroll
      for (int r = 0; r < 8; ++r)
        ob[(size_t)(gr0 + r) * kDim] = acc[mt][r] + bias;
    }
  }
}

// ---------------------------------------------------------------------------
extern "C" void kernel_launch(void* const* d_in, const int* in_sizes, int n_in,
                              void* d_out, int out_size, void* d_ws,
                              size_t ws_size, hipStream_t stream) {
  (void)in_sizes; (void)n_in; (void)out_size; (void)ws_size;
  const float* x       = (const float*)d_in[0];
  const float* qkv_w   = (const float*)d_in[1];
  const float* qkv_b   = (const float*)d_in[2];
  const float* proj_w  = (const float*)d_in[3];
  const float* proj_b  = (const float*)d_in[4];
  const float* rpb     = (const float*)d_in[5];
  const int*   rel_idx = (const int*)d_in[6];
  float* out = (float*)d_out;

  // Workspace: swizzled weights (384K + 128K bf16) + swizzled bias (128K f32).
  char* ws = (char*)d_ws;
  __bf16* wqkv_sw  = (__bf16*)ws;
  __bf16* wproj_sw = (__bf16*)(ws + 768 * 256 * sizeof(__bf16));
  float*  bias_sw  = (float*)(ws + (768 + 256) * 256 * sizeof(__bf16));

  swizzle_w_kernel<<<(768 * 256 + 255) / 256, 256, 0, stream>>>(
      qkv_w, wqkv_sw, 768, 256);
  swizzle_w_kernel<<<(256 * 256 + 255) / 256, 256, 0, stream>>>(
      proj_w, wproj_sw, 256, 256);
  swizzle_bias_kernel<<<(32768 + 255) / 256, 256, 0, stream>>>(
      rpb, rel_idx, bias_sw);

  // 81920 bf16 = 163840 B dynamic LDS (within the 320 KB WGP limit).
  fused_win_kernel<<<kNWin, 256, 163840, stream>>>(
      x, wqkv_sw, qkv_b, wproj_sw, proj_b, bias_sw, out);
}
